// VNect_84293028151425
// MI455X (gfx1250) — matrix operand order repaired and checked by reading, not compiled, not run on hardware
//
#include <hip/hip_runtime.h>
#include <math.h>

// ---------------------------------------------------------------------------
// VNect-style network on MI455X (gfx1250).
// Implicit-GEMM convs via v_wmma_f32_16x16x32_f16 with:
//   * 4-wave workgroups: block tile = 64 oc x 64 px
//   * activation K-tiles staged to LDS with global_load_async_to_lds_b128,
//     double-buffered, synced with s_wait_asynccnt + workgroup barriers
//   * B-fragments read from LDS (ds_load_b128), A-fragments from global
//   * hot loop uses incremental (kb,kh,kw) counters -- no div/mod per step
// Activations are f16 NHWC so fragments are contiguous 16B chunks.
// ---------------------------------------------------------------------------

typedef _Float16 v8h  __attribute__((ext_vector_type(8)));
typedef _Float16 v16h __attribute__((ext_vector_type(16)));
typedef float    v8f  __attribute__((ext_vector_type(8)));

union V16 { v16h v; v8h h[2]; };

#define MODE_RELU 1
#define MODE_BN   2
#define MODE_RES  4
#define MODE_F32  8

// async copy of one 16-byte chunk: global (64-bit addr) -> LDS (32-bit addr)
__device__ __forceinline__ void async_b128(uint32_t lds_addr, uint64_t gaddr)
{
    asm volatile("global_load_async_to_lds_b128 %0, %1, off"
                 :: "v"(lds_addr), "v"(gaddr) : "memory");
}

// ---------------------------------------------------------------------------
// Same-padding conv (KHW = 1 or 3).
//   act : [img*P + pixel][Cin] (f16 NHWC)   wt : [oc][kh][kw][ic] (f16)
// Block: 128 threads (4 waves). Wave w owns oc-subtile blockIdx.y*64 + w*16
// and computes all 4 pixel subtiles (64 px). Grid: x = Ntot/64, y = CoutPad/64.
// ---------------------------------------------------------------------------
__global__ __launch_bounds__(128)
void conv_wmma_k(const _Float16* __restrict__ act,
                 const _Float16* __restrict__ wt,
                 const _Float16* __restrict__ zbuf,   // 256B of zeros (pad reads)
                 int Cin, int H, int W, int P, int KHW, int pad,
                 _Float16* __restrict__ out16, float* __restrict__ out32,
                 int outStride, int outChanOff, int clipM,
                 const float* __restrict__ bias,
                 const float* __restrict__ bnScale,
                 const float* __restrict__ bnShift,
                 const _Float16* __restrict__ resid, int residStride,
                 int mode)
{
    __shared__ _Float16 tileLds[2][64 * 32];     // 2 x 4KB: [pixel][k]

    const int tid  = threadIdx.x;
    const int lane = tid & 31;
    const int wid  = tid >> 5;
    const int N0 = blockIdx.x * 64;
    const int M0 = blockIdx.y * 64 + wid * 16;

    const int S = KHW * KHW * (Cin >> 5);        // total K steps
    const int Kstride = KHW * KHW * Cin;

    // ---- loader mapping: thread -> two pixel rows (nA, nB), 8-half chunk ----
    const int ko = (tid & 3) * 8;                // half offset within K row
    const int nA = tid >> 2;                     // rows 0..31
    const int nB = nA + 32;                      // rows 32..63
    const int gpA = N0 + nA, gpB = N0 + nB;
    const int imgA = gpA / P, ppA = gpA - imgA * P;
    const int imgB = gpB / P, ppB = gpB - imgB * P;
    const int ohA = ppA / W, owA = ppA - ohA * W;
    const int ohB = ppB / W, owB = ppB - ohB * W;

    const uint32_t ldsBase = (uint32_t)(uintptr_t)(&tileLds[0][0]);
    const uint64_t zaddr   = (uint64_t)(uintptr_t)zbuf;

    // issue-stream position: (kbI within tap, tap decomposed as khI,kwI)
    int kbI = 0, khI = 0, kwI = 0;
    auto issue_step = [&](int buf) {
        const uint32_t lbase = ldsBase + (uint32_t)buf * 4096u;
        {   // row nA
            const int hh = ohA + khI - pad, ww = owA + kwI - pad;
            const bool ok = (hh >= 0) && (hh < H) && (ww >= 0) && (ww < W);
            const uint64_t g = ok
                ? (uint64_t)(uintptr_t)(act + ((size_t)imgA * P + (size_t)hh * W + ww) * Cin + kbI + ko)
                : zaddr;
            async_b128(lbase + (uint32_t)(nA * 32 + ko) * 2u, g);
        }
        {   // row nB
            const int hh = ohB + khI - pad, ww = owB + kwI - pad;
            const bool ok = (hh >= 0) && (hh < H) && (ww >= 0) && (ww < W);
            const uint64_t g = ok
                ? (uint64_t)(uintptr_t)(act + ((size_t)imgB * P + (size_t)hh * W + ww) * Cin + kbI + ko)
                : zaddr;
            async_b128(lbase + (uint32_t)(nB * 32 + ko) * 2u, g);
        }
        // advance (no div/mod)
        kbI += 32;
        if (kbI == Cin) {
            kbI = 0;
            if (++kwI == KHW) { kwI = 0; ++khI; }
        }
    };

    // ---- compute-side fragment indexing ----
    const int arow  = M0 + (lane & 15);
    const int akoff = (lane < 16) ? 0 : 8;       // A: K {0..7,16..23} / {8..15,24..31}
    const _Float16* aPtr = wt + (size_t)arow * Kstride + akoff;  // advances 32/step
    const int col   = lane & 15;
    const int bkoff = (lane < 16) ? 0 : 8;       // B: K {0..7,16..23} / {8..15,24..31}

    v8f acc[4];
    #pragma unroll
    for (int u = 0; u < 4; ++u)
        #pragma unroll
        for (int i = 0; i < 8; ++i) acc[u][i] = 0.0f;

    issue_step(0);
    for (int s = 0; s < S; ++s) {
        const int buf = s & 1;
        if (s + 1 < S) {
            issue_step(buf ^ 1);
            asm volatile("s_wait_asynccnt 0x2" ::: "memory");
        } else {
            asm volatile("s_wait_asynccnt 0x0" ::: "memory");
        }
        __syncthreads();                          // tile for step s ready

        V16 a;
        a.h[0] = *(const v8h*)(aPtr);
        a.h[1] = *(const v8h*)(aPtr + 16);
        aPtr += 32;                               // t*Cin + kb is monotone +32
        #pragma unroll
        for (int u = 0; u < 4; ++u) {
            V16 b;
            const v8h* bp = (const v8h*)&tileLds[buf][(u * 16 + col) * 32 + bkoff];
            b.h[0] = bp[0];
            b.h[1] = bp[2];
            acc[u] = __builtin_amdgcn_wmma_f32_16x16x32_f16(
                         false, a.v, false, b.v, (short)0, acc[u], false, false);
        }
        __syncthreads();                          // everyone done before overwrite
    }

    // ---- epilogue ----
    const int row0 = (lane < 16) ? 0 : 8;
    #pragma unroll
    for (int u = 0; u < 4; ++u) {
        const int n = N0 + u * 16 + col;
        #pragma unroll
        for (int r = 0; r < 8; ++r) {
            const int oc = M0 + row0 + r;
            if (oc >= clipM) continue;
            float v = acc[u][r];
            if (mode & MODE_BN) v = v * bnScale[oc] + bnShift[oc];
            else                v += bias[oc];
            if (mode & MODE_RES) v += (float)resid[(size_t)n * residStride + oc];
            if (mode & MODE_RELU) v = fmaxf(v, 0.0f);
            const size_t oidx = (size_t)n * outStride + outChanOff + oc;
            if (mode & MODE_F32) out32[oidx] = v;
            else                 out16[oidx] = (_Float16)v;
        }
    }
}

// ---------------------------------------------------------------------------
// Transposed conv 23x23 -> 46x46, kernel 4x4, stride 2, pad 2 (16 taps with
// parity masks). Same 4-wave / LDS-async structure as conv_wmma_k.
//   act : [img*529 + pixel][Cin]   wt : [oc][kh*4+kw][Cin] (flipped)
// ---------------------------------------------------------------------------
__global__ __launch_bounds__(128)
void deconv_wmma_k(const _Float16* __restrict__ act,
                   const _Float16* __restrict__ wt,
                   const _Float16* __restrict__ zbuf,
                   int Cin,
                   _Float16* __restrict__ out16,
                   int outStride, int outChanOff, int clipM,
                   const float* __restrict__ bias, int mode)
{
    __shared__ _Float16 tileLds[2][64 * 32];

    const int tid  = threadIdx.x;
    const int lane = tid & 31;
    const int wid  = tid >> 5;
    const int N0 = blockIdx.x * 64;
    const int M0 = blockIdx.y * 64 + wid * 16;

    const int S = 16 * (Cin >> 5);
    const int Kstride = 16 * Cin;

    const int ko = (tid & 3) * 8;
    const int nA = tid >> 2;
    const int nB = nA + 32;
    const int gpA = N0 + nA, gpB = N0 + nB;
    const int imgA = gpA / 2116, ppA = gpA - imgA * 2116;
    const int imgB = gpB / 2116, ppB = gpB - imgB * 2116;
    const int ohA = ppA / 46, owA = ppA - ohA * 46;
    const int ohB = ppB / 46, owB = ppB - ohB * 46;

    const uint32_t ldsBase = (uint32_t)(uintptr_t)(&tileLds[0][0]);
    const uint64_t zaddr   = (uint64_t)(uintptr_t)zbuf;

    int kbI = 0, khI = 0, kwI = 0;
    auto issue_step = [&](int buf) {
        const uint32_t lbase = ldsBase + (uint32_t)buf * 4096u;
        {
            const int dh = ohA + khI - 2, dw = owA + kwI - 2;
            const bool ok = (dh >= 0) && (dw >= 0) && !(dh & 1) && !(dw & 1) &&
                            ((dh >> 1) < 23) && ((dw >> 1) < 23);
            const uint64_t g = ok
                ? (uint64_t)(uintptr_t)(act + ((size_t)imgA * 529 + (size_t)(dh >> 1) * 23 + (dw >> 1)) * Cin + kbI + ko)
                : zaddr;
            async_b128(lbase + (uint32_t)(nA * 32 + ko) * 2u, g);
        }
        {
            const int dh = ohB + khI - 2, dw = owB + kwI - 2;
            const bool ok = (dh >= 0) && (dw >= 0) && !(dh & 1) && !(dw & 1) &&
                            ((dh >> 1) < 23) && ((dw >> 1) < 23);
            const uint64_t g = ok
                ? (uint64_t)(uintptr_t)(act + ((size_t)imgB * 529 + (size_t)(dh >> 1) * 23 + (dw >> 1)) * Cin + kbI + ko)
                : zaddr;
            async_b128(lbase + (uint32_t)(nB * 32 + ko) * 2u, g);
        }
        kbI += 32;
        if (kbI == Cin) {
            kbI = 0;
            if (++kwI == 4) { kwI = 0; ++khI; }
        }
    };

    const int arow  = M0 + (lane & 15);
    const int akoff = (lane < 16) ? 0 : 8;
    const _Float16* aPtr = wt + (size_t)arow * Kstride + akoff;
    const int col   = lane & 15;
    const int bkoff = (lane < 16) ? 0 : 8;

    v8f acc[4];
    #pragma unroll
    for (int u = 0; u < 4; ++u)
        #pragma unroll
        for (int i = 0; i < 8; ++i) acc[u][i] = 0.0f;

    issue_step(0);
    for (int s = 0; s < S; ++s) {
        const int buf = s & 1;
        if (s + 1 < S) {
            issue_step(buf ^ 1);
            asm volatile("s_wait_asynccnt 0x2" ::: "memory");
        } else {
            asm volatile("s_wait_asynccnt 0x0" ::: "memory");
        }
        __syncthreads();

        V16 a;
        a.h[0] = *(const v8h*)(aPtr);
        a.h[1] = *(const v8h*)(aPtr + 16);
        aPtr += 32;
        #pragma unroll
        for (int u = 0; u < 4; ++u) {
            V16 b;
            const v8h* bp = (const v8h*)&tileLds[buf][(u * 16 + col) * 32 + bkoff];
            b.h[0] = bp[0];
            b.h[1] = bp[2];
            acc[u] = __builtin_amdgcn_wmma_f32_16x16x32_f16(
                         false, a.v, false, b.v, (short)0, acc[u], false, false);
        }
        __syncthreads();
    }

    const int row0 = (lane < 16) ? 0 : 8;
    #pragma unroll
    for (int u = 0; u < 4; ++u) {
        const int n = N0 + u * 16 + col;
        #pragma unroll
        for (int r = 0; r < 8; ++r) {
            const int oc = M0 + row0 + r;
            if (oc >= clipM) continue;
            float v = acc[u][r] + bias[oc];
            if (mode & MODE_RELU) v = fmaxf(v, 0.0f);
            out16[(size_t)n * outStride + outChanOff + oc] = (_Float16)v;
        }
    }
}

// ------------------------- prep / small kernels ----------------------------

__global__ void zero16_k(_Float16* __restrict__ p, int n)
{
    int i = blockIdx.x * blockDim.x + threadIdx.x;
    if (i < n) p[i] = (_Float16)0;
}

// x (64,1024,23,23) f32 NCHW -> f16 NHWC [b*529+p][1024]
__global__ void cvt_x_k(const float* __restrict__ x, _Float16* __restrict__ x16)
{
    size_t idx = (size_t)blockIdx.x * blockDim.x + threadIdx.x;
    const size_t total = (size_t)64 * 1024 * 529;
    if (idx >= total) return;
    int p = (int)(idx % 529);
    size_t t = idx / 529;
    int c = (int)(t % 1024);
    int b = (int)(t / 1024);
    x16[((size_t)b * 529 + p) * 1024 + c] = (_Float16)x[idx];
}

__global__ void w1x1_k(const float* __restrict__ w, _Float16* __restrict__ o, int total)
{
    int i = blockIdx.x * blockDim.x + threadIdx.x;
    if (i < total) o[i] = (_Float16)w[i];
}

// copy with zero-padded tail (row-padded [oc][ic] weights)
__global__ void w1x1pad_k(const float* __restrict__ w, _Float16* __restrict__ o,
                          int realTotal, int padTotal)
{
    int i = blockIdx.x * blockDim.x + threadIdx.x;
    if (i >= padTotal) return;
    o[i] = (i < realTotal) ? (_Float16)w[i] : (_Float16)0;
}

// (O,I,3,3) OIHW -> [oc][t][ic]
__global__ void w3x3_k(const float* __restrict__ w, _Float16* __restrict__ o, int O, int I)
{
    int idx = blockIdx.x * blockDim.x + threadIdx.x;
    int total = O * I * 9;
    if (idx >= total) return;
    int t = idx % 9;
    int r = idx / 9;
    int ic = r % I;
    int oc = r / I;
    o[(size_t)oc * 9 * I + (size_t)t * I + ic] = (_Float16)w[idx];
}

// f1_w (128,212,3,3) -> [oc][t*224 + ic], ic zero-padded 212..223
__global__ void wf1_k(const float* __restrict__ w, _Float16* __restrict__ o)
{
    int idx = blockIdx.x * blockDim.x + threadIdx.x;
    int total = 128 * 9 * 224;
    if (idx >= total) return;
    int oc = idx / (9 * 224);
    int r  = idx % (9 * 224);
    int t  = r / 224;
    int ic = r % 224;
    float v = 0.0f;
    if (ic < 212) v = w[((size_t)oc * 212 + ic) * 9 + t];
    o[idx] = (_Float16)v;
}

// deconv weights (Ic, OcReal, 4, 4) -> [oc][t][ic] with spatial flip, oc padded
__global__ void wdeconv_k(const float* __restrict__ w, _Float16* __restrict__ o,
                          int Ic, int OcReal, int OcPad)
{
    int idx = blockIdx.x * blockDim.x + threadIdx.x;
    int total = OcPad * 16 * Ic;
    if (idx >= total) return;
    int oc = idx / (16 * Ic);
    int r  = idx % (16 * Ic);
    int t  = r / Ic;
    int ic = r % Ic;
    int kh = t >> 2, kw = t & 3;
    float v = 0.0f;
    if (oc < OcReal)
        v = w[(((size_t)ic * OcReal + oc) * 4 + (3 - kh)) * 4 + (3 - kw)];
    o[idx] = (_Float16)v;
}

// fold conv-bias + batchnorm into scale/shift
__global__ void bnprep_k(const float* g, const float* be, const float* m,
                         const float* v, const float* b,
                         float* scale, float* shift, int C)
{
    int i = blockIdx.x * blockDim.x + threadIdx.x;
    if (i >= C) return;
    float s = g[i] * rsqrtf(v[i] + 1e-5f);
    scale[i] = s;
    shift[i] = (b[i] - m[i]) * s + be[i];
}

// out3 = sqrt(o1[3j]^2 + o1[3j+1]^2 + o1[3j+2]^2) into cat[191+j]; zero pad 212..223
__global__ void mag_k(_Float16* __restrict__ cat, int NP2)
{
    int idx = blockIdx.x * blockDim.x + threadIdx.x;
    int total = NP2 * 33;
    if (idx >= total) return;
    int p = idx / 33;
    int j = idx % 33;
    _Float16* row = cat + (size_t)p * 224;
    if (j < 21) {
        float a = (float)row[3 * j], b = (float)row[3 * j + 1], c = (float)row[3 * j + 2];
        row[191 + j] = (_Float16)sqrtf(a * a + b * b + c * c);
    } else {
        row[212 + (j - 21)] = (_Float16)0;
    }
}

// Final argmax + gather. ff2: [b][pixel(hh*46+ww)][112] f32.
// Reference reshape (B,112,46,46)->(B,4,46,46,28): for (a,w_,h_,k):
//   frem = w_*1288 + h_*28 + k ; cc = frem/2116 ; r2 = frem%2116
//   hh = r2/46 ; ww = r2%46 ; channel = a*28+cc
__global__ void final_k(const float* __restrict__ ff2, float* __restrict__ out)
{
    int b = blockIdx.x;
    int k = threadIdx.x;
    if (k >= 28) return;
    const float* fb = ff2 + (size_t)b * 2116 * 112;

    float best = -3.4e38f;
    int bw = 0, bh = 0;
    for (int h_ = 0; h_ < 46; ++h_) {
        float mm = -3.4e38f;
        int mw = 0;
        for (int w_ = 0; w_ < 46; ++w_) {
            int frem = w_ * 1288 + h_ * 28 + k;
            int cc = frem / 2116;
            int r2 = frem - cc * 2116;
            int hh = r2 / 46, ww = r2 - hh * 46;
            float v = fb[(size_t)(hh * 46 + ww) * 112 + cc];   // a = 0 (heatmap)
            if (v > mm) { mm = v; mw = w_; }                    // first-occurrence
        }
        if (mm > best) { best = mm; bh = h_; bw = mw; }         // first-occurrence
    }
    int frem = bw * 1288 + bh * 28 + k;
    int cc = frem / 2116;
    int r2 = frem - cc * 2116;
    int hh = r2 / 46, ww = r2 - hh * 46;
    #pragma unroll
    for (int a = 1; a <= 3; ++a)
        out[(size_t)b * 84 + k * 3 + (a - 1)] =
            fb[(size_t)(hh * 46 + ww) * 112 + a * 28 + cc];
}

// ---------------------------------------------------------------------------

extern "C" void kernel_launch(void* const* d_in, const int* in_sizes, int n_in,
                              void* d_out, int out_size, void* d_ws, size_t ws_size,
                              hipStream_t stream)
{
    (void)in_sizes; (void)n_in; (void)out_size; (void)ws_size;

    const float* x    = (const float*)d_in[0];
    const float* w1   = (const float*)d_in[1];
    const float* b1   = (const float*)d_in[2];
    const float* g1   = (const float*)d_in[3];
    const float* be1  = (const float*)d_in[4];
    const float* m1   = (const float*)d_in[5];
    const float* v1   = (const float*)d_in[6];
    const float* w2   = (const float*)d_in[7];
    const float* b2   = (const float*)d_in[8];
    const float* g2   = (const float*)d_in[9];
    const float* be2  = (const float*)d_in[10];
    const float* m2   = (const float*)d_in[11];
    const float* v2   = (const float*)d_in[12];
    const float* w3   = (const float*)d_in[13];
    const float* b3   = (const float*)d_in[14];
    const float* wb1  = (const float*)d_in[15];
    const float* bb1  = (const float*)d_in[16];
    const float* wb2  = (const float*)d_in[17];
    const float* bb2  = (const float*)d_in[18];
    const float* wb3  = (const float*)d_in[19];
    const float* bb3  = (const float*)d_in[20];
    const float* wc1  = (const float*)d_in[21];
    const float* bc1  = (const float*)d_in[22];
    const float* wc2  = (const float*)d_in[23];
    const float* bc2  = (const float*)d_in[24];
    const float* wf1  = (const float*)d_in[25];
    const float* bf1  = (const float*)d_in[26];
    const float* wf2  = (const float*)d_in[27];
    const float* bf2  = (const float*)d_in[28];

    const int NP1 = 64 * 529;     // 23x23 pixels, all batches
    const int NP2 = 64 * 2116;    // 46x46 pixels, all batches
    const int NB1 = NP1 / 64;     // 529  pixel-blocks (64 px)
    const int NB2 = NP2 / 64;     // 2116 pixel-blocks

    // ---- workspace carve-up (bump allocator, 256B aligned) ----
    char* ws = (char*)d_ws;
    size_t off = 0;
    auto alloc = [&](size_t bytes) -> char* {
        char* p = ws + off;
        off += (bytes + 255) & ~(size_t)255;
        return p;
    };

    _Float16* ZB  = (_Float16*)alloc(256);                     // zero scratch
    _Float16* W1  = (_Float16*)alloc((size_t)512 * 1024 * 2);
    _Float16* W2  = (_Float16*)alloc((size_t)512 * 4608 * 2);
    _Float16* W3  = (_Float16*)alloc((size_t)1024 * 512 * 2);
    _Float16* WB1 = (_Float16*)alloc((size_t)256 * 1024 * 2);
    _Float16* WB2 = (_Float16*)alloc((size_t)128 * 2304 * 2);
    _Float16* WB3 = (_Float16*)alloc((size_t)256 * 128 * 2);
    _Float16* WC1 = (_Float16*)alloc((size_t)64 * 16 * 256 * 2);
    _Float16* WC2 = (_Float16*)alloc((size_t)128 * 16 * 256 * 2);
    _Float16* WF1 = (_Float16*)alloc((size_t)128 * 9 * 224 * 2);
    _Float16* WF2 = (_Float16*)alloc((size_t)128 * 128 * 2);   // padded 112->128 rows
    float* SC1 = (float*)alloc(512 * 4);
    float* SH1 = (float*)alloc(512 * 4);
    float* SC2 = (float*)alloc(512 * 4);
    float* SH2 = (float*)alloc(512 * 4);

    _Float16* X16 = (_Float16*)alloc((size_t)NP1 * 1024 * 2);  // 69.3 MB
    _Float16* H1  = (_Float16*)alloc((size_t)NP1 * 512 * 2);   // 34.7 MB
    _Float16* H2  = (_Float16*)alloc((size_t)NP1 * 512 * 2);   // 34.7 MB
    _Float16* H3  = (_Float16*)alloc((size_t)NP1 * 1024 * 2);  // 69.3 MB

    // region reuse (lifetimes disjoint):
    _Float16* HB1 = H1;              // 256ch @ 23x23
    _Float16* HB2 = H2;              // 128ch @ 23x23
    _Float16* HB3 = X16;             // 256ch @ 23x23 (x dead after residual)
    _Float16* CAT = H3;              // 224ch @ 46x46 (h3 dead after b1)
    _Float16* HF1 = H2;              // 128ch @ 46x46
    float*    FF2 = (float*)X16;     // 112ch @ 46x46 f32 (b3 dead after deconvs)

    const int TB = 256;

    // ---- prep ----
    zero16_k<<<1, 128, 0, stream>>>(ZB, 128);
    {
        size_t n = (size_t)64 * 1024 * 529;
        cvt_x_k<<<dim3((unsigned)((n + TB - 1) / TB)), TB, 0, stream>>>(x, X16);
    }
    w1x1_k<<<(512 * 1024 + TB - 1) / TB, TB, 0, stream>>>(w1, W1, 512 * 1024);
    w3x3_k<<<(512 * 512 * 9 + TB - 1) / TB, TB, 0, stream>>>(w2, W2, 512, 512);
    w1x1_k<<<(1024 * 512 + TB - 1) / TB, TB, 0, stream>>>(w3, W3, 1024 * 512);
    w1x1_k<<<(256 * 1024 + TB - 1) / TB, TB, 0, stream>>>(wb1, WB1, 256 * 1024);
    w3x3_k<<<(128 * 256 * 9 + TB - 1) / TB, TB, 0, stream>>>(wb2, WB2, 128, 256);
    w1x1_k<<<(256 * 128 + TB - 1) / TB, TB, 0, stream>>>(wb3, WB3, 256 * 128);
    wdeconv_k<<<(64 * 16 * 256 + TB - 1) / TB, TB, 0, stream>>>(wc1, WC1, 256, 63, 64);
    wdeconv_k<<<(128 * 16 * 256 + TB - 1) / TB, TB, 0, stream>>>(wc2, WC2, 256, 128, 128);
    wf1_k<<<(128 * 9 * 224 + TB - 1) / TB, TB, 0, stream>>>(wf1, WF1);
    w1x1pad_k<<<(128 * 128 + TB - 1) / TB, TB, 0, stream>>>(wf2, WF2, 112 * 128, 128 * 128);
    bnprep_k<<<2, TB, 0, stream>>>(g1, be1, m1, v1, b1, SC1, SH1, 512);
    bnprep_k<<<2, TB, 0, stream>>>(g2, be2, m2, v2, b2, SC2, SH2, 512);

    // ---- residual block ----
    conv_wmma_k<<<dim3(NB1, 8), 128, 0, stream>>>(           // 1x1 1024->512 BN+ReLU
        X16, W1, ZB, 1024, 23, 23, 529, 1, 0,
        H1, nullptr, 512, 0, 512, b1, SC1, SH1, nullptr, 0, MODE_BN | MODE_RELU);
    conv_wmma_k<<<dim3(NB1, 8), 128, 0, stream>>>(           // 3x3 512->512 BN+ReLU
        H1, W2, ZB, 512, 23, 23, 529, 3, 1,
        H2, nullptr, 512, 0, 512, b2, SC2, SH2, nullptr, 0, MODE_BN | MODE_RELU);
    conv_wmma_k<<<dim3(NB1, 16), 128, 0, stream>>>(          // 1x1 512->1024 + x
        H2, W3, ZB, 512, 23, 23, 529, 1, 0,
        H3, nullptr, 1024, 0, 1024, b3, nullptr, nullptr, X16, 1024, MODE_RES);

    // ---- bottleneck ----
    conv_wmma_k<<<dim3(NB1, 4), 128, 0, stream>>>(
        H3, WB1, ZB, 1024, 23, 23, 529, 1, 0,
        HB1, nullptr, 256, 0, 256, bb1, nullptr, nullptr, nullptr, 0, 0);
    conv_wmma_k<<<dim3(NB1, 2), 128, 0, stream>>>(
        HB1, WB2, ZB, 256, 23, 23, 529, 3, 1,
        HB2, nullptr, 128, 0, 128, bb2, nullptr, nullptr, nullptr, 0, 0);
    conv_wmma_k<<<dim3(NB1, 4), 128, 0, stream>>>(
        HB2, WB3, ZB, 128, 23, 23, 529, 1, 0,
        HB3, nullptr, 256, 0, 256, bb3, nullptr, nullptr, nullptr, 0, 0);

    // ---- deconvs into the concat buffer (224ch, 46x46) ----
    deconv_wmma_k<<<dim3(NB2, 1), 128, 0, stream>>>(
        HB3, WC1, ZB, 256, CAT, 224, 0, 63, bc1, 0);             // out1: ch 0..62
    deconv_wmma_k<<<dim3(NB2, 2), 128, 0, stream>>>(
        HB3, WC2, ZB, 256, CAT, 224, 63, 128, bc2, MODE_RELU);   // out2: ch 63..190
    mag_k<<<(NP2 * 33 + TB - 1) / TB, TB, 0, stream>>>(CAT, NP2);

    // ---- head ----
    conv_wmma_k<<<dim3(NB2, 2), 128, 0, stream>>>(
        CAT, WF1, ZB, 224, 46, 46, 2116, 3, 1,
        HF1, nullptr, 128, 0, 128, bf1, nullptr, nullptr, nullptr, 0, MODE_RELU);
    conv_wmma_k<<<dim3(NB2, 2), 128, 0, stream>>>(
        HF1, WF2, ZB, 128, 46, 46, 2116, 1, 0,
        nullptr, FF2, 112, 0, 112, bf2, nullptr, nullptr, nullptr, 0, MODE_F32);

    // ---- argmax + gather -> (64, 84) ----
    final_k<<<64, 32, 0, stream>>>(FF2, (float*)d_out);
}
